// CausalSelfAttention_3229815407331
// MI455X (gfx1250) — compile-verified, matrix operand order checked by reading, not run on hardware
//
#include <hip/hip_runtime.h>
#include <hip/hip_bf16.h>

#define S_LEN   2048
#define DMODEL  1024
#define NHEADS  16
#define HDIM    64
#define BATCH   4
#define BS      (BATCH * S_LEN)   // 8192 rows

typedef __bf16 bf16_t;
typedef __attribute__((ext_vector_type(16))) __bf16       v16bf;
typedef __attribute__((ext_vector_type(8)))  float        v8f;
typedef __attribute__((ext_vector_type(4)))  unsigned int u32x4;
typedef __attribute__((ext_vector_type(4)))  int          i32x4;

#define AS1 __attribute__((address_space(1)))
#define AS3 __attribute__((address_space(3)))

#if defined(__AMDGCN__) && \
    __has_builtin(__builtin_amdgcn_global_load_async_to_lds_b128) && \
    __has_builtin(__builtin_amdgcn_s_wait_asynccnt)
#define USE_ASYNC_LDS 1
#else
#define USE_ASYNC_LDS 0
#endif

union Frag16 { v16bf v; u32x4 u[2]; };

// D = A(16x32 bf16) * B(32x16 bf16) + C(16x16 f32)
__device__ __forceinline__ v8f wmma_bf16(v16bf a, v16bf b, v8f c) {
  return __builtin_amdgcn_wmma_f32_16x16x32_bf16(
      /*neg_a=*/false, a, /*neg_b=*/false, b,
      /*c_mod=*/(short)0, c, /*reuse_a=*/false, /*reuse_b=*/false);
}

// 16-bit fragment layout (ISA 7.12.2): lanes 0-15 hold row (A) / col (B) = lane,
// VGPR0-3 = K k0+8*half .. +7, VGPR4-7 = K k0+16+8*half .. +7.
// Each lane: two contiguous 16-byte loads from its row. Works for global or LDS.
__device__ __forceinline__ v16bf load_frag(const bf16_t* row, int k0, int hf) {
  Frag16 f;
  f.u[0] = *(const u32x4*)(row + k0 + 8 * hf);
  f.u[1] = *(const u32x4*)(row + k0 + 16 + 8 * hf);
  return f.v;
}

// reduce across the 16 lanes of a half-wave (xor masks < 16 stay in-half)
__device__ __forceinline__ float rowmax16(float v) {
  v = fmaxf(v, __shfl_xor(v, 1, 32));
  v = fmaxf(v, __shfl_xor(v, 2, 32));
  v = fmaxf(v, __shfl_xor(v, 4, 32));
  v = fmaxf(v, __shfl_xor(v, 8, 32));
  return v;
}
__device__ __forceinline__ float rowsum16(float v) {
  v += __shfl_xor(v, 1, 32);
  v += __shfl_xor(v, 2, 32);
  v += __shfl_xor(v, 4, 32);
  v += __shfl_xor(v, 8, 32);
  return v;
}

// ---------------- conversion kernels ----------------

__global__ void cvt_bf16_kernel(const float* __restrict__ in,
                                bf16_t* __restrict__ out, int n) {
  int idx = (blockIdx.x * blockDim.x + threadIdx.x) * 4;
  if (idx + 3 < n) {
    float4 f = *(const float4*)(in + idx);
    out[idx + 0] = (bf16_t)f.x;
    out[idx + 1] = (bf16_t)f.y;
    out[idx + 2] = (bf16_t)f.z;
    out[idx + 3] = (bf16_t)f.w;
  }
}

// W is [K=1024, N=1024] row-major (x @ W); produce WT[N,K] bf16 so GEMM
// B-fragments are contiguous along K.
__global__ void cvt_wT_kernel(const float* __restrict__ W,
                              bf16_t* __restrict__ WT) {
  int tid = blockIdx.x * blockDim.x + threadIdx.x;   // 0 .. 1M-1
  int k = tid >> 10;
  int n = tid & (DMODEL - 1);
  WT[(size_t)n * DMODEL + k] = (bf16_t)W[(size_t)k * DMODEL + n];
}

// ---------------- projection GEMM -> head-major bf16 ----------------
// Register-blocked: each wave computes a 16(M) x 64(N) strip: one A fragment
// reused against 4 B fragments per k-step, 4 f32 accumulators.
// Out[B,H,S,Hd] = split_heads(X @ W + bias), X bf16 [8192,1024], WT bf16 [N,K]
__global__ void __launch_bounds__(256)
gemm_qkv_kernel(const bf16_t* __restrict__ X, const bf16_t* __restrict__ WT,
                const float* __restrict__ bias, bf16_t* __restrict__ Out) {
  int lane = threadIdx.x & 31;
  int wave = threadIdx.x >> 5;
  int hf = lane >> 4;
  int ln = lane & 15;
  int nBase = blockIdx.x * 64;
  int mBase = (blockIdx.y * 8 + wave) * 16;

  const bf16_t* arow = X + (size_t)(mBase + ln) * DMODEL;
  const bf16_t* brow0 = WT + (size_t)(nBase + 0 * 16 + ln) * DMODEL;
  const bf16_t* brow1 = WT + (size_t)(nBase + 1 * 16 + ln) * DMODEL;
  const bf16_t* brow2 = WT + (size_t)(nBase + 2 * 16 + ln) * DMODEL;
  const bf16_t* brow3 = WT + (size_t)(nBase + 3 * 16 + ln) * DMODEL;

  v8f acc0 = {}, acc1 = {}, acc2 = {}, acc3 = {};
#pragma unroll 4
  for (int k0 = 0; k0 < DMODEL; k0 += 32) {
    v16bf a = load_frag(arow, k0, hf);
    acc0 = wmma_bf16(a, load_frag(brow0, k0, hf), acc0);
    acc1 = wmma_bf16(a, load_frag(brow1, k0, hf), acc1);
    acc2 = wmma_bf16(a, load_frag(brow2, k0, hf), acc2);
    acc3 = wmma_bf16(a, load_frag(brow3, k0, hf), acc3);
  }

  int h = nBase >> 6;   // nBase is a multiple of 64 -> one head per wave strip
#pragma unroll
  for (int r = 0; r < 8; ++r) {
    int m = mBase + r + 8 * hf;               // C/D layout: row = r + 8*half
    int b_ = m >> 11, s = m & (S_LEN - 1);
    size_t rowbase = (((size_t)(b_ * NHEADS + h)) * S_LEN + s) * HDIM;
    Out[rowbase + 0 * 16 + ln] = (bf16_t)(acc0[r] + bias[nBase + 0 * 16 + ln]);
    Out[rowbase + 1 * 16 + ln] = (bf16_t)(acc1[r] + bias[nBase + 1 * 16 + ln]);
    Out[rowbase + 2 * 16 + ln] = (bf16_t)(acc2[r] + bias[nBase + 2 * 16 + ln]);
    Out[rowbase + 3 * 16 + ln] = (bf16_t)(acc3[r] + bias[nBase + 3 * 16 + ln]);
  }
}

// ---------------- fused causal flash attention ----------------
// grid (S/64, B*H), 128 threads (4 waves). Each wave owns 16 query rows.
__global__ void __launch_bounds__(128)
attn_fwd_kernel(const bf16_t* __restrict__ Q, const bf16_t* __restrict__ K,
                const bf16_t* __restrict__ V, bf16_t* __restrict__ O) {
  __shared__ __align__(16) bf16_t Kt[32 * 64];      // [key][hd]
  __shared__ __align__(16) bf16_t Vt[64 * 32];      // [hd][key] (transposed)
  __shared__ __align__(16) bf16_t Pt[4][16 * 32];   // per-wave P tile [q][key]

  int lane = threadIdx.x & 31;
  int wave = threadIdx.x >> 5;
  int hf = lane >> 4;
  int ln = lane & 15;
  int bh = blockIdx.y;               // b*H + h
  int qBase = blockIdx.x * 64;
  int waveQ = qBase + wave * 16;

  const bf16_t* Qb = Q + (size_t)bh * S_LEN * HDIM;
  const bf16_t* Kb = K + (size_t)bh * S_LEN * HDIM;
  const bf16_t* Vb = V + (size_t)bh * S_LEN * HDIM;

  // Q fragments for hd 0..31 / 32..63, resident all kernel
  const bf16_t* qrowp = Qb + (size_t)(waveQ + ln) * HDIM;
  v16bf aq0 = load_frag(qrowp, 0, hf);
  v16bf aq1 = load_frag(qrowp, 32, hf);

  v8f acc0 = {}, acc1 = {}, acc2 = {}, acc3 = {};
  float mrun[8], lrun[8];
#pragma unroll
  for (int r = 0; r < 8; ++r) { mrun[r] = -3.0e38f; lrun[r] = 0.0f; }

  int nkb = (qBase >> 5) + 2;        // key blocks of 32 covering [0, qBase+64)
  int tkey = threadIdx.x >> 2;       // 0..31
  int tc = (threadIdx.x & 3) * 16;   // hd chunk

  for (int kb = 0; kb < nkb; ++kb) {
    int keyBase = kb * 32;
    // ---- staging: K tile via gfx1250 async-to-LDS when available ----
    const bf16_t* krow = Kb + (size_t)(keyBase + tkey) * HDIM + tc;
#if USE_ASYNC_LDS
    __builtin_amdgcn_global_load_async_to_lds_b128(
        (AS1 i32x4*)krow, (AS3 i32x4*)&Kt[tkey * 64 + tc], 0, 0);
    __builtin_amdgcn_global_load_async_to_lds_b128(
        (AS1 i32x4*)(krow + 8), (AS3 i32x4*)&Kt[tkey * 64 + tc + 8], 0, 0);
#else
    *(u32x4*)&Kt[tkey * 64 + tc] = *(const u32x4*)(krow);
    *(u32x4*)&Kt[tkey * 64 + tc + 8] = *(const u32x4*)(krow + 8);
#endif
    // V tile: load to registers, scatter-transpose into LDS
    {
      const bf16_t* vrow = Vb + (size_t)(keyBase + tkey) * HDIM + tc;
      bf16_t tmp[16];
      *(u32x4*)&tmp[0] = *(const u32x4*)(vrow);
      *(u32x4*)&tmp[8] = *(const u32x4*)(vrow + 8);
#pragma unroll
      for (int i = 0; i < 16; ++i) Vt[(tc + i) * 32 + tkey] = tmp[i];
    }
#if USE_ASYNC_LDS
    __builtin_amdgcn_s_wait_asynccnt(0);
#endif
    __syncthreads();

    // scores: S0 = Q @ K[0..15]^T, S1 = Q @ K[16..31]^T  (hd is the WMMA K-dim)
    v8f s0 = {}, s1 = {};
    s0 = wmma_bf16(aq0, load_frag(&Kt[ln * 64], 0, hf), s0);
    s0 = wmma_bf16(aq1, load_frag(&Kt[ln * 64], 32, hf), s0);
    s1 = wmma_bf16(aq0, load_frag(&Kt[(16 + ln) * 64], 0, hf), s1);
    s1 = wmma_bf16(aq1, load_frag(&Kt[(16 + ln) * 64], 32, hf), s1);

    int key0 = keyBase + ln;
    int key1 = keyBase + 16 + ln;
#pragma unroll
    for (int r = 0; r < 8; ++r) {
      int qrow = waveQ + r + 8 * hf;
      bool ok0 = key0 <= qrow, ok1 = key1 <= qrow;
      float v0 = ok0 ? s0[r] * 0.125f : -3.0e38f;   // 1/sqrt(64)
      float v1 = ok1 ? s1[r] * 0.125f : -3.0e38f;
      float mx = rowmax16(fmaxf(v0, v1));
      float mnew = fmaxf(mrun[r], mx);
      float scl = __expf(mrun[r] - mnew);
      float p0 = ok0 ? __expf(v0 - mnew) : 0.0f;
      float p1 = ok1 ? __expf(v1 - mnew) : 0.0f;
      float rs = rowsum16(p0 + p1);
      lrun[r] = lrun[r] * scl + rs;
      mrun[r] = mnew;
      acc0[r] *= scl; acc1[r] *= scl; acc2[r] *= scl; acc3[r] *= scl;
      // C-layout -> A-layout conversion via LDS round trip (same-wave DS order)
      int prow = (r + 8 * hf) * 32;
      Pt[wave][prow + ln] = (bf16_t)p0;
      Pt[wave][prow + 16 + ln] = (bf16_t)p1;
    }
    // P(16x32) @ V(32x64): B fragments contiguous along key in transposed Vt
    v16bf ap = load_frag(&Pt[wave][ln * 32], 0, hf);
    acc0 = wmma_bf16(ap, load_frag(&Vt[(0 * 16 + ln) * 32], 0, hf), acc0);
    acc1 = wmma_bf16(ap, load_frag(&Vt[(1 * 16 + ln) * 32], 0, hf), acc1);
    acc2 = wmma_bf16(ap, load_frag(&Vt[(2 * 16 + ln) * 32], 0, hf), acc2);
    acc3 = wmma_bf16(ap, load_frag(&Vt[(3 * 16 + ln) * 32], 0, hf), acc3);
    __syncthreads();
  }

  // epilogue: normalize and merge heads -> [B,S,D] bf16
  int h = bh & (NHEADS - 1);
  int b_ = bh >> 4;
#pragma unroll
  for (int r = 0; r < 8; ++r) {
    float inv = 1.0f / lrun[r];
    int qrow = waveQ + r + 8 * hf;
    size_t base = ((size_t)(b_ * S_LEN + qrow)) * DMODEL + h * HDIM;
    O[base + 0 * 16 + ln] = (bf16_t)(acc0[r] * inv);
    O[base + 1 * 16 + ln] = (bf16_t)(acc1[r] * inv);
    O[base + 2 * 16 + ln] = (bf16_t)(acc2[r] * inv);
    O[base + 3 * 16 + ln] = (bf16_t)(acc3[r] * inv);
  }
}

// ---------------- output projection GEMM -> fp32 ----------------
// Same 16x64-per-wave register blocking, fp32 row-major store to d_out.
__global__ void __launch_bounds__(256)
gemm_out_kernel(const bf16_t* __restrict__ X, const bf16_t* __restrict__ WT,
                const float* __restrict__ bias, float* __restrict__ Out) {
  int lane = threadIdx.x & 31;
  int wave = threadIdx.x >> 5;
  int hf = lane >> 4;
  int ln = lane & 15;
  int nBase = blockIdx.x * 64;
  int mBase = (blockIdx.y * 8 + wave) * 16;

  const bf16_t* arow = X + (size_t)(mBase + ln) * DMODEL;
  const bf16_t* brow0 = WT + (size_t)(nBase + 0 * 16 + ln) * DMODEL;
  const bf16_t* brow1 = WT + (size_t)(nBase + 1 * 16 + ln) * DMODEL;
  const bf16_t* brow2 = WT + (size_t)(nBase + 2 * 16 + ln) * DMODEL;
  const bf16_t* brow3 = WT + (size_t)(nBase + 3 * 16 + ln) * DMODEL;

  v8f acc0 = {}, acc1 = {}, acc2 = {}, acc3 = {};
#pragma unroll 4
  for (int k0 = 0; k0 < DMODEL; k0 += 32) {
    v16bf a = load_frag(arow, k0, hf);
    acc0 = wmma_bf16(a, load_frag(brow0, k0, hf), acc0);
    acc1 = wmma_bf16(a, load_frag(brow1, k0, hf), acc1);
    acc2 = wmma_bf16(a, load_frag(brow2, k0, hf), acc2);
    acc3 = wmma_bf16(a, load_frag(brow3, k0, hf), acc3);
  }

  float b0 = bias[nBase + 0 * 16 + ln];
  float b1 = bias[nBase + 1 * 16 + ln];
  float b2 = bias[nBase + 2 * 16 + ln];
  float b3 = bias[nBase + 3 * 16 + ln];
#pragma unroll
  for (int r = 0; r < 8; ++r) {
    int m = mBase + r + 8 * hf;
    size_t rowbase = (size_t)m * DMODEL + nBase;
    Out[rowbase + 0 * 16 + ln] = acc0[r] + b0;
    Out[rowbase + 1 * 16 + ln] = acc1[r] + b1;
    Out[rowbase + 2 * 16 + ln] = acc2[r] + b2;
    Out[rowbase + 3 * 16 + ln] = acc3[r] + b3;
  }
}

extern "C" void kernel_launch(void* const* d_in, const int* in_sizes, int n_in,
                              void* d_out, int out_size, void* d_ws, size_t ws_size,
                              hipStream_t stream) {
  (void)in_sizes; (void)n_in; (void)out_size; (void)ws_size;
  const float* x  = (const float*)d_in[0];
  const float* Wq = (const float*)d_in[1];
  const float* bq = (const float*)d_in[2];
  const float* Wk = (const float*)d_in[3];
  const float* bk = (const float*)d_in[4];
  const float* Wv = (const float*)d_in[5];
  const float* bv = (const float*)d_in[6];
  const float* Wo = (const float*)d_in[7];
  const float* bo = (const float*)d_in[8];
  float* out = (float*)d_out;

  char* ws = (char*)d_ws;
  const size_t MB = 1024 * 1024;
  bf16_t* xb  = (bf16_t*)(ws + 0);        // 16 MB  x bf16
  bf16_t* WqT = (bf16_t*)(ws + 16 * MB);  //  2 MB each
  bf16_t* WkT = (bf16_t*)(ws + 18 * MB);
  bf16_t* WvT = (bf16_t*)(ws + 20 * MB);
  bf16_t* WoT = (bf16_t*)(ws + 22 * MB);
  bf16_t* Qh  = (bf16_t*)(ws + 24 * MB);  // 16 MB each [B,H,S,Hd]
  bf16_t* Kh  = (bf16_t*)(ws + 40 * MB);
  bf16_t* Vh  = (bf16_t*)(ws + 56 * MB);
  bf16_t* Ah  = (bf16_t*)(ws + 72 * MB);  // 16 MB attn out [B,S,D] bf16

  // 1) convert activations + transpose-convert weights to bf16
  cvt_bf16_kernel<<<(BS * DMODEL) / (256 * 4), 256, 0, stream>>>(x, xb, BS * DMODEL);
  cvt_wT_kernel<<<(DMODEL * DMODEL) / 256, 256, 0, stream>>>(Wq, WqT);
  cvt_wT_kernel<<<(DMODEL * DMODEL) / 256, 256, 0, stream>>>(Wk, WkT);
  cvt_wT_kernel<<<(DMODEL * DMODEL) / 256, 256, 0, stream>>>(Wv, WvT);
  cvt_wT_kernel<<<(DMODEL * DMODEL) / 256, 256, 0, stream>>>(Wo, WoT);

  // 2) Q/K/V projections (WMMA, 16x64 per wave) into head-major layout
  dim3 ggrid(DMODEL / 64, BS / 128);
  gemm_qkv_kernel<<<ggrid, 256, 0, stream>>>(xb, WqT, bq, Qh);
  gemm_qkv_kernel<<<ggrid, 256, 0, stream>>>(xb, WkT, bk, Kh);
  gemm_qkv_kernel<<<ggrid, 256, 0, stream>>>(xb, WvT, bv, Vh);

  // 3) fused causal flash attention (WMMA + async LDS staging + online softmax)
  attn_fwd_kernel<<<dim3(S_LEN / 64, BATCH * NHEADS), 128, 0, stream>>>(Qh, Kh, Vh, Ah);

  // 4) output projection -> fp32 d_out
  gemm_out_kernel<<<ggrid, 256, 0, stream>>>(Ah, WoT, bo, out);
}
